// SchNetInteraction_46497315946644
// MI455X (gfx1250) — matrix operand order repaired and checked by reading, not compiled, not run on hardware
//
#include <hip/hip_runtime.h>
#include <hip/hip_bf16.h>

typedef __bf16 bf16_t;
typedef __attribute__((ext_vector_type(16))) __bf16 v16bf;
typedef __attribute__((ext_vector_type(8)))  float  v8f;

static constexpr int NN = 32768;   // nodes
static constexpr int HH = 128;     // hidden
static constexpr int GG = 50;      // gaussians
static constexpr int EE = 524288;  // edges
static constexpr float LOG2F_C = 0.69314718056f;
static constexpr float PI_F = 3.14159265358979f;

// ---------------- device helpers ----------------

__device__ __forceinline__ v8f wmma_bf16(v16bf a, v16bf b, v8f c) {
  // D = A(16x32 bf16) x B(32x16 bf16) + C(16x16 f32)
  return __builtin_amdgcn_wmma_f32_16x16x32_bf16(
      /*neg_a=*/false, a, /*neg_b=*/false, b,
      /*c_mod=*/(short)0, c, /*reuse_a=*/false, /*reuse_b=*/false);
}

__device__ __forceinline__ float sspf(float x) {
  // shifted softplus: log(1+exp(x)) - log(2), numerically stable
  float ax = fabsf(x);
  return fmaxf(x, 0.f) + __logf(1.f + __expf(-ax)) - LOG2F_C;
}

__device__ __forceinline__ int f2ord(float f) {
  int i = __float_as_int(f);
  return (i < 0) ? (i ^ 0x7fffffff) : i;
}
__device__ __forceinline__ float ord2f(int o) {
  return __int_as_float((o < 0) ? (o ^ 0x7fffffff) : o);
}

// A-fragment (16x32 bf16) per ISA layout: lane m=l&15, g=l>>4;
// elem e -> vgpr v=e>>1, pair p=e&1; k = (v>=4?16:0) + g*8 + (v&3)*2 + p
__device__ __forceinline__ v16bf load_a_frag(const bf16_t* tile, int stride,
                                             int kc, int lane) {
  int m = lane & 15, g = lane >> 4;
  const bf16_t* rp = tile + m * stride + kc * 32;
  v16bf a;
#pragma unroll
  for (int e = 0; e < 16; ++e) {
    int v = e >> 1, p = e & 1;
    int kl = ((v >= 4) ? 16 : 0) + g * 8 + (v & 3) * 2 + p;
    a[e] = rp[kl];
  }
  return a;
}

// ---------------- kernels ----------------

// init softmax state + zero agg
__global__ void init_kernel(int* __restrict__ mOrd, float* __restrict__ sDen,
                            float* __restrict__ agg) {
  size_t i = (size_t)blockIdx.x * blockDim.x + threadIdx.x;
  if (i < (size_t)NN) {
    mOrd[i] = f2ord(-__builtin_inff());
    sDen[i] = 0.f;
  }
  if (i < (size_t)NN * 128) agg[i] = 0.f;
}

// pack fp32 weight [Kreal x 128] (row stride 128) into WMMA B-fragment order:
// frag (kc, nt): 32 lanes x 16 bf16; lane col n = nt*16+(l&15); k = kc*32+(e>=8?16:0)+(l>>4)*8+(e&7)
__global__ void pack_b_kernel(const float* __restrict__ w, bf16_t* __restrict__ dst,
                              int Kreal, int kchunks) {
  int tid = blockIdx.x * blockDim.x + threadIdx.x;
  int total = kchunks * 8 * 32;
  if (tid >= total) return;
  int lane = tid & 31, frag = tid >> 5;
  int kc = frag >> 3, nt = frag & 7;
  int g = lane >> 4, n = nt * 16 + (lane & 15);
  bf16_t* out = dst + (size_t)frag * 512 + lane * 16;
#pragma unroll
  for (int e = 0; e < 16; ++e) {
    int k = kc * 32 + ((e >= 8) ? 16 : 0) + g * 8 + (e & 7);
    float f = (k < Kreal) ? w[(size_t)k * 128 + n] : 0.f;
    out[e] = (bf16_t)f;
  }
}

// P1 = h@W1a, P2 = h@W1b, X1 = h@lin1_w   (three N x 128 x 128 GEMMs)
__global__ __launch_bounds__(128) void node_pre_kernel(
    const float* __restrict__ h, const bf16_t* __restrict__ pW1a,
    const bf16_t* __restrict__ pW1b, const bf16_t* __restrict__ pLin1,
    float* __restrict__ P1, float* __restrict__ P2, float* __restrict__ X1) {
  __shared__ bf16_t sA[4][16][128];
  int lane = threadIdx.x & 31, wave = threadIdx.x >> 5;
  size_t node0 = ((size_t)blockIdx.x * 4 + wave) * 16;
  const float* hp = h + node0 * 128;
  for (int i = lane; i < 16 * 128; i += 32)
    sA[wave][i >> 7][i & 127] = (bf16_t)hp[i];
  __syncthreads();
  v16bf aF[4];
#pragma unroll
  for (int kc = 0; kc < 4; ++kc)
    aF[kc] = load_a_frag(&sA[wave][0][0], 128, kc, lane);
  int m = lane & 15, g = lane >> 4;
  const bf16_t* Ws[3] = {pW1a, pW1b, pLin1};
  float* Os[3] = {P1, P2, X1};
#pragma unroll
  for (int w = 0; w < 3; ++w) {
    const v16bf* bp = (const v16bf*)Ws[w];
    float* op = Os[w] + node0 * 128;
#pragma unroll
    for (int nt = 0; nt < 8; ++nt) {
      v8f acc = {};
#pragma unroll
      for (int kc = 0; kc < 4; ++kc)
        acc = wmma_bf16(aF[kc], bp[(kc * 8 + nt) * 32 + lane], acc);
      int n = nt * 16 + m;
#pragma unroll
      for (int v = 0; v < 8; ++v) op[(size_t)(g * 8 + v) * 128 + n] = acc[v];
    }
  }
}

// attention score per edge + segment max (ordered-int atomicMax)
__global__ __launch_bounds__(128) void edge_attn_kernel(
    const float* __restrict__ pos, const int* __restrict__ row_idx,
    const int* __restrict__ col_idx, const unsigned char* __restrict__ mask,
    const float* __restrict__ P1, const float* __restrict__ P2,
    const bf16_t* __restrict__ pW1c, const float* __restrict__ b1,
    const float* __restrict__ w2, const float* __restrict__ b2,
    float* __restrict__ score, int* __restrict__ mOrd) {
  __shared__ int sRow[4][16], sCol[4][16];
  __shared__ float sD[4][16], sSc[4][16];
  __shared__ bf16_t sEA[4][16][64];
  int lane = threadIdx.x & 31, wave = threadIdx.x >> 5;
  size_t e0 = ((size_t)blockIdx.x * 4 + wave) * 16;
  if (lane < 16) {
    int r = row_idx[e0 + lane], c = col_idx[e0 + lane];
    sRow[wave][lane] = r;
    sCol[wave][lane] = c;
    float dx = pos[r * 3 + 0] - pos[c * 3 + 0];
    float dy = pos[r * 3 + 1] - pos[c * 3 + 1];
    float dz = pos[r * 3 + 2] - pos[c * 3 + 2];
    sD[wave][lane] = sqrtf(dx * dx + dy * dy + dz * dz);
    sSc[wave][lane] = 0.f;
  }
  __syncthreads();
  const float delta = 10.0f / 49.0f;
  const float coeff = -0.5f / (delta * delta);
  for (int i = lane; i < 16 * 64; i += 32) {
    int mm = i >> 6, gg = i & 63;
    float val = 0.f;
    if (gg < GG) {
      float t = sD[wave][mm] - delta * (float)gg;
      val = __expf(coeff * t * t);
    }
    sEA[wave][mm][gg] = (bf16_t)val;
  }
  __syncthreads();
  v16bf aF0 = load_a_frag(&sEA[wave][0][0], 64, 0, lane);
  v16bf aF1 = load_a_frag(&sEA[wave][0][0], 64, 1, lane);
  int m = lane & 15, g = lane >> 4;
  const v16bf* bp = (const v16bf*)pW1c;
  float partial[8];
#pragma unroll
  for (int v = 0; v < 8; ++v) partial[v] = 0.f;
#pragma unroll
  for (int nt = 0; nt < 8; ++nt) {
    v8f acc = {};
    acc = wmma_bf16(aF0, bp[nt * 32 + lane], acc);
    acc = wmma_bf16(aF1, bp[(8 + nt) * 32 + lane], acc);
    int n = nt * 16 + m;
    float b1n = b1[n], w2n = w2[n];
#pragma unroll
    for (int v = 0; v < 8; ++v) {
      int er = g * 8 + v;
      float hid = acc[v] + P1[(size_t)sRow[wave][er] * 128 + n] +
                  P2[(size_t)sCol[wave][er] * 128 + n] + b1n;
      partial[v] += fmaxf(hid, 0.f) * w2n;
    }
  }
#pragma unroll
  for (int v = 0; v < 8; ++v) atomicAdd(&sSc[wave][g * 8 + v], partial[v]);
  __syncthreads();
  if (lane < 16) {
    float sc = sSc[wave][lane] + b2[0];
    score[e0 + lane] = sc;
    float neg = mask[e0 + lane] ? sc : -__builtin_inff();
    atomicMax(&mOrd[sRow[wave][lane]], f2ord(neg));
  }
}

// e = mask ? exp(score - m) : 0 ; segment sum
__global__ void softmax_den_kernel(const float* __restrict__ score,
                                   const int* __restrict__ row_idx,
                                   const unsigned char* __restrict__ mask,
                                   const int* __restrict__ mOrd,
                                   float* __restrict__ expE,
                                   float* __restrict__ sDen) {
  size_t e = (size_t)blockIdx.x * blockDim.x + threadIdx.x;
  if (e >= (size_t)EE) return;
  int r = row_idx[e];
  float mf = ord2f(mOrd[r]);
  if (!isfinite(mf)) mf = 0.f;
  float ex = mask[e] ? __expf(score[e] - mf) : 0.f;
  expE[e] = ex;
  if (ex != 0.f) atomicAdd(&sDen[r], ex);
}

// filter MLP on decay-scaled smearing, msg = X1[row]*W, scatter-add into agg[col]
__global__ __launch_bounds__(128) void edge_msg_kernel(
    const float* __restrict__ pos, const int* __restrict__ row_idx,
    const int* __restrict__ col_idx, const unsigned char* __restrict__ mask,
    const float* __restrict__ expE, const float* __restrict__ sDen,
    const bf16_t* __restrict__ pFw1, const float* __restrict__ fb1,
    const bf16_t* __restrict__ pFw2, const float* __restrict__ fb2,
    const float* __restrict__ X1, float* __restrict__ agg) {
  __shared__ int sRow[4][16], sCol[4][16];
  __shared__ float sD[4][16], sDec[4][16], sC[4][16];
  __shared__ bf16_t sEA[4][16][64];
  __shared__ bf16_t sT[4][16][128];
  int lane = threadIdx.x & 31, wave = threadIdx.x >> 5;
  size_t e0 = ((size_t)blockIdx.x * 4 + wave) * 16;
  if (lane < 16) {
    int r = row_idx[e0 + lane], c = col_idx[e0 + lane];
    sRow[wave][lane] = r;
    sCol[wave][lane] = c;
    float dx = pos[r * 3 + 0] - pos[c * 3 + 0];
    float dy = pos[r * 3 + 1] - pos[c * 3 + 1];
    float dz = pos[r * 3 + 2] - pos[c * 3 + 2];
    float d = sqrtf(dx * dx + dy * dy + dz * dz);
    sD[wave][lane] = d;
    sC[wave][lane] = 0.5f * (__cosf(d * (PI_F / 10.0f)) + 1.0f);
    sDec[wave][lane] =
        mask[e0 + lane] ? expE[e0 + lane] / (sDen[r] + 1e-16f) : 1.0f;
  }
  __syncthreads();
  const float delta = 10.0f / 49.0f;
  const float coeff = -0.5f / (delta * delta);
  for (int i = lane; i < 16 * 64; i += 32) {
    int mm = i >> 6, gg = i & 63;
    float val = 0.f;
    if (gg < GG) {
      float t = sD[wave][mm] - delta * (float)gg;
      val = __expf(coeff * t * t) * sDec[wave][mm];  // fold decay in
    }
    sEA[wave][mm][gg] = (bf16_t)val;
  }
  __syncthreads();
  v16bf aF0 = load_a_frag(&sEA[wave][0][0], 64, 0, lane);
  v16bf aF1 = load_a_frag(&sEA[wave][0][0], 64, 1, lane);
  int m = lane & 15, g = lane >> 4;
  const v16bf* bp1 = (const v16bf*)pFw1;
#pragma unroll
  for (int nt = 0; nt < 8; ++nt) {
    v8f acc = {};
    acc = wmma_bf16(aF0, bp1[nt * 32 + lane], acc);
    acc = wmma_bf16(aF1, bp1[(8 + nt) * 32 + lane], acc);
    int n = nt * 16 + m;
    float b = fb1[n];
#pragma unroll
    for (int v = 0; v < 8; ++v)
      sT[wave][g * 8 + v][n] = (bf16_t)sspf(acc[v] + b);
  }
  __syncthreads();
  v16bf aT[4];
#pragma unroll
  for (int kc = 0; kc < 4; ++kc)
    aT[kc] = load_a_frag(&sT[wave][0][0], 128, kc, lane);
  const v16bf* bp2 = (const v16bf*)pFw2;
#pragma unroll
  for (int nt = 0; nt < 8; ++nt) {
    v8f acc = {};
#pragma unroll
    for (int kc = 0; kc < 4; ++kc)
      acc = wmma_bf16(aT[kc], bp2[(kc * 8 + nt) * 32 + lane], acc);
    int n = nt * 16 + m;
    float b = fb2[n];
#pragma unroll
    for (int v = 0; v < 8; ++v) {
      int er = g * 8 + v;
      float Wv = (acc[v] + b) * sC[wave][er];
      float msg = X1[(size_t)sRow[wave][er] * 128 + n] * Wv;
      atomicAdd(&agg[(size_t)sCol[wave][er] * 128 + n], msg);
    }
  }
}

// h_update = ssp(agg@lin2 + b2) @ lin + b
__global__ __launch_bounds__(128) void node_out_kernel(
    const float* __restrict__ agg, const bf16_t* __restrict__ pLin2,
    const float* __restrict__ lin2_b, const bf16_t* __restrict__ pLin,
    const float* __restrict__ lin_b, float* __restrict__ out) {
  __shared__ bf16_t sA[4][16][128];
  __shared__ bf16_t sT[4][16][128];
  int lane = threadIdx.x & 31, wave = threadIdx.x >> 5;
  size_t node0 = ((size_t)blockIdx.x * 4 + wave) * 16;
  const float* ap = agg + node0 * 128;
  for (int i = lane; i < 16 * 128; i += 32)
    sA[wave][i >> 7][i & 127] = (bf16_t)ap[i];
  __syncthreads();
  v16bf aF[4];
#pragma unroll
  for (int kc = 0; kc < 4; ++kc)
    aF[kc] = load_a_frag(&sA[wave][0][0], 128, kc, lane);
  int m = lane & 15, g = lane >> 4;
  const v16bf* bp2 = (const v16bf*)pLin2;
#pragma unroll
  for (int nt = 0; nt < 8; ++nt) {
    v8f acc = {};
#pragma unroll
    for (int kc = 0; kc < 4; ++kc)
      acc = wmma_bf16(aF[kc], bp2[(kc * 8 + nt) * 32 + lane], acc);
    int n = nt * 16 + m;
    float b = lin2_b[n];
#pragma unroll
    for (int v = 0; v < 8; ++v)
      sT[wave][g * 8 + v][n] = (bf16_t)sspf(acc[v] + b);
  }
  __syncthreads();
  v16bf aT[4];
#pragma unroll
  for (int kc = 0; kc < 4; ++kc)
    aT[kc] = load_a_frag(&sT[wave][0][0], 128, kc, lane);
  const v16bf* bpL = (const v16bf*)pLin;
#pragma unroll
  for (int nt = 0; nt < 8; ++nt) {
    v8f acc = {};
#pragma unroll
    for (int kc = 0; kc < 4; ++kc)
      acc = wmma_bf16(aT[kc], bpL[(kc * 8 + nt) * 32 + lane], acc);
    int n = nt * 16 + m;
    float b = lin_b[n];
#pragma unroll
    for (int v = 0; v < 8; ++v)
      out[(node0 + g * 8 + v) * 128 + n] = acc[v] + b;
  }
}

// ---------------- host launch ----------------

extern "C" void kernel_launch(void* const* d_in, const int* in_sizes, int n_in,
                              void* d_out, int out_size, void* d_ws,
                              size_t ws_size, hipStream_t stream) {
  (void)in_sizes; (void)n_in; (void)out_size; (void)ws_size;
  const float* h        = (const float*)d_in[0];
  const float* pos      = (const float*)d_in[1];
  const int*   eidx     = (const int*)d_in[2];
  const unsigned char* mask = (const unsigned char*)d_in[3];
  const float* attn_w1  = (const float*)d_in[4];
  const float* attn_b1  = (const float*)d_in[5];
  const float* attn_w2  = (const float*)d_in[6];
  const float* attn_b2  = (const float*)d_in[7];
  const float* filt_w1  = (const float*)d_in[8];
  const float* filt_b1  = (const float*)d_in[9];
  const float* filt_w2  = (const float*)d_in[10];
  const float* filt_b2  = (const float*)d_in[11];
  const float* lin1_w   = (const float*)d_in[12];
  const float* lin2_w   = (const float*)d_in[13];
  const float* lin2_b   = (const float*)d_in[14];
  const float* lin_w    = (const float*)d_in[15];
  const float* lin_b    = (const float*)d_in[16];
  float* out = (float*)d_out;

  const int* row_idx = eidx;
  const int* col_idx = eidx + EE;

  char* ws = (char*)d_ws;
  auto alloc = [&](size_t bytes) -> char* {
    char* p = ws;
    ws += (bytes + 255) & ~(size_t)255;
    return p;
  };
  bf16_t* pW1a  = (bf16_t*)alloc(4 * 8 * 512 * 2);
  bf16_t* pW1b  = (bf16_t*)alloc(4 * 8 * 512 * 2);
  bf16_t* pW1c  = (bf16_t*)alloc(2 * 8 * 512 * 2);
  bf16_t* pFw1  = (bf16_t*)alloc(2 * 8 * 512 * 2);
  bf16_t* pFw2  = (bf16_t*)alloc(4 * 8 * 512 * 2);
  bf16_t* pLin1 = (bf16_t*)alloc(4 * 8 * 512 * 2);
  bf16_t* pLin2 = (bf16_t*)alloc(4 * 8 * 512 * 2);
  bf16_t* pLin  = (bf16_t*)alloc(4 * 8 * 512 * 2);
  float* P1    = (float*)alloc((size_t)NN * 128 * 4);
  float* P2    = (float*)alloc((size_t)NN * 128 * 4);
  float* X1    = (float*)alloc((size_t)NN * 128 * 4);
  float* agg   = (float*)alloc((size_t)NN * 128 * 4);
  float* score = (float*)alloc((size_t)EE * 4);
  float* expE  = (float*)alloc((size_t)EE * 4);
  float* sDen  = (float*)alloc((size_t)NN * 4);
  int*   mOrd  = (int*)alloc((size_t)NN * 4);

  // 1. init softmax state + zero agg
  init_kernel<<<(NN * 128 + 255) / 256, 256, 0, stream>>>(mOrd, sDen, agg);

  // 2. pack weights into WMMA B-fragment order (bf16)
  pack_b_kernel<<<4, 256, 0, stream>>>(attn_w1,            pW1a, 128, 4);
  pack_b_kernel<<<4, 256, 0, stream>>>(attn_w1 + 128*128,  pW1b, 128, 4);
  pack_b_kernel<<<2, 256, 0, stream>>>(attn_w1 + 256*128,  pW1c, GG, 2);
  pack_b_kernel<<<2, 256, 0, stream>>>(filt_w1,            pFw1, GG, 2);
  pack_b_kernel<<<4, 256, 0, stream>>>(filt_w2,            pFw2, 128, 4);
  pack_b_kernel<<<4, 256, 0, stream>>>(lin1_w,             pLin1, 128, 4);
  pack_b_kernel<<<4, 256, 0, stream>>>(lin2_w,             pLin2, 128, 4);
  pack_b_kernel<<<4, 256, 0, stream>>>(lin_w,              pLin, 128, 4);

  // 3. node-level pre-GEMMs: P1, P2, X1
  node_pre_kernel<<<NN / 64, 128, 0, stream>>>(h, pW1a, pW1b, pLin1, P1, P2, X1);

  // 4. edge attention scores + segment max
  edge_attn_kernel<<<EE / 64, 128, 0, stream>>>(pos, row_idx, col_idx, mask, P1,
                                                P2, pW1c, attn_b1, attn_w2,
                                                attn_b2, score, mOrd);

  // 5. softmax numerators + denominators
  softmax_den_kernel<<<EE / 256, 256, 0, stream>>>(score, row_idx, mask, mOrd,
                                                   expE, sDen);

  // 6. CFConv filter + message scatter
  edge_msg_kernel<<<EE / 64, 128, 0, stream>>>(pos, row_idx, col_idx, mask,
                                               expE, sDen, pFw1, filt_b1, pFw2,
                                               filt_b2, X1, agg);

  // 7. output node MLP
  node_out_kernel<<<NN / 64, 128, 0, stream>>>(agg, pLin2, lin2_b, pLin, lin_b,
                                               out);

  // 8. second tuple output: pos passthrough
  hipMemcpyAsync(out + (size_t)NN * 128, pos, (size_t)NN * 3 * sizeof(float),
                 hipMemcpyDeviceToDevice, stream);
}
